// SlidingWindowAttention_25477746000076
// MI455X (gfx1250) — compile-verified, matrix operand order checked by reading, not run on hardware
//
#include <hip/hip_runtime.h>
#include <hip/hip_bf16.h>

typedef unsigned short u16;
typedef unsigned int   u32;
typedef unsigned long long u64;
typedef __bf16 bf16;
typedef __attribute__((ext_vector_type(16))) __bf16 v16bf;
typedef __attribute__((ext_vector_type(8)))  float  v8f;
typedef __attribute__((ext_vector_type(4)))  u32    v4u;
typedef __attribute__((ext_vector_type(8)))  int    v8i;
typedef __attribute__((ext_vector_type(4)))  int    v4i;

#if defined(__has_builtin)
#if __has_builtin(__builtin_amdgcn_tensor_load_to_lds) && __has_builtin(__builtin_amdgcn_s_wait_tensorcnt)
#define USE_TDM 1
#endif
#if __has_builtin(__builtin_amdgcn_permlane16)
#define USE_PERMLANE 1
#endif
#endif

union FragU { v16bf v; uint4 q[2]; };

__device__ __forceinline__ u16 f2bf(float x) {
  u32 u = __float_as_uint(x);
  u32 r = u + 0x7FFFu + ((u >> 16) & 1u);   // round-to-nearest-even
  return (u16)(r >> 16);
}

__device__ __forceinline__ v8f wmma_bf16(v16bf a, v16bf b, v8f c) {
  return __builtin_amdgcn_wmma_f32_16x16x32_bf16(false, a, false, b, (short)0, c, false, false);
}

// ---- 16-lane xor butterfly via v_permlane16_b32 (VALU, no LDS pipe) ----
__host__ __device__ constexpr u32 sel_xor(int base, int k) {
  u32 s = 0;
  for (int i = 0; i < 8; i++) s |= (u32)(((base + i) ^ k) & 0xF) << (4 * i);
  return s;
}

template <int K16>
__device__ __forceinline__ float lane_xor16(float x) {
#if defined(USE_PERMLANE)
  constexpr u32 s0 = sel_xor(0, K16);
  constexpr u32 s1 = sel_xor(8, K16);
  int v = __float_as_int(x);
  return __int_as_float(__builtin_amdgcn_permlane16(v, v, (int)s0, (int)s1, false, false));
#else
  return __shfl_xor(x, K16, 32);
#endif
}

__device__ __forceinline__ float redmax16(float x) {
  x = fmaxf(x, lane_xor16<1>(x));
  x = fmaxf(x, lane_xor16<2>(x));
  x = fmaxf(x, lane_xor16<4>(x));
  x = fmaxf(x, lane_xor16<8>(x));
  return x;
}
__device__ __forceinline__ float redsum16(float x) {
  x += lane_xor16<1>(x);
  x += lane_xor16<2>(x);
  x += lane_xor16<4>(x);
  x += lane_xor16<8>(x);
  return x;
}

// A/B fragment load from row-major bf16 (16 rows x 32 K):
// lanes 0-15: row = lane, K {0..7, 16..23}; lanes 16-31: row = lane-16, K {8..15, 24..31}
__device__ __forceinline__ v16bf ld_frag(const u16* p0, int stride, int lane) {
  const u16* p = p0 + (size_t)(lane & 15) * stride + ((lane & 16) ? 8 : 0);
  FragU f;
  f.q[0] = *(const uint4*)(p);
  f.q[1] = *(const uint4*)(p + 16);
  return f.v;
}

__device__ __forceinline__ v16bf ld_frag_clamp(const u16* base, int row0, int stride,
                                               int col, int lane, int maxRow) {
  int r = row0 + (lane & 15);
  r = r < 0 ? 0 : (r > maxRow ? maxRow : r);
  const u16* p = base + (size_t)r * stride + col + ((lane & 16) ? 8 : 0);
  FragU f;
  f.q[0] = *(const uint4*)(p);
  f.q[1] = *(const uint4*)(p + 16);
  return f.v;
}

// V as B-fragment: lane n holds column (col), keys follow A-style K pattern (strided loads)
__device__ __forceinline__ v16bf ld_vfrag(const u16* Vbase, int k0, int col, int lane) {
  int kb = (lane & 16) ? 8 : 0;
  v16bf v;
#pragma unroll
  for (int e = 0; e < 16; e++) {
    int kk = kb + (e < 8 ? e : e + 8);   // {0..7,16..23} or {8..15,24..31}
    int row = k0 + kk;
    row = row < 0 ? 0 : (row > 2047 ? 2047 : row);
    v[e] = *(const bf16*)(Vbase + (size_t)row * 1024 + col);
  }
  return v;
}

#if defined(USE_TDM)
// Issue a TDM 2D tile load: tile (tile_d0 elems x tile_d1 rows), bf16, row stride = strideElems.
// D# layout per CDNA5 ISA ch.8 (group0: count/lds/global/type; group1: dims/tile/strides).
// 6-arg builtin form: (g0, g1, g2, g3, g_extra, cpol).
__device__ __forceinline__ void tdm_load_2d(u32 ldsAddr, u64 gAddr, u32 tensor_d0, u32 tensor_d1,
                                            u32 tile_d0, u32 tile_d1, u64 strideElems) {
  v4u g0 = (v4u){1u,                       // count=1, is_restore=0, gather off
                 ldsAddr,                  // lds_addr [63:32]
                 (u32)gAddr,               // global_addr lo
                 ((u32)(gAddr >> 32) & 0x01FFFFFFu) | 0x80000000u};  // addr hi | type=2
  v8i g1 = (v8i){(int)(1u << 16),                          // data_size=1 (2 bytes)
                 (int)((tensor_d0 & 0xFFFFu) << 16),       // tensor_dim0 [79:48] lo16
                 (int)(((tensor_d0 >> 16) & 0xFFFFu) | ((tensor_d1 & 0xFFFFu) << 16)),
                 (int)(((tensor_d1 >> 16) & 0xFFFFu) | (tile_d0 << 16)),   // tile_dim0 [127:112]
                 (int)(tile_d1 & 0xFFFFu),                 // tile_dim1 [143:128], tile_dim2=0
                 (int)(u32)(strideElems & 0xFFFFFFFFu),    // tensor_dim0_stride lo
                 (int)(u32)((strideElems >> 32) & 0xFFFFu),
                 0};
  v4i z4 = (v4i){0, 0, 0, 0};
  v8i z8 = (v8i){0, 0, 0, 0, 0, 0, 0, 0};
  __builtin_amdgcn_tensor_load_to_lds(g0, g1, z4, z4, z8, 0);
}
#endif

// ---------------- conversion kernels ----------------
__global__ __launch_bounds__(256) void f32_to_bf16_kernel(const float* __restrict__ in,
                                                          u16* __restrict__ out, int n) {
  int i = blockIdx.x * 256 + threadIdx.x;
  if (i < n) out[i] = f2bf(in[i]);
}

// W [K=1024][N=1024] f32  ->  Wt [N][K] bf16
__global__ __launch_bounds__(256) void transpose_to_bf16_kernel(const float* __restrict__ in,
                                                                u16* __restrict__ out) {
  int idx = blockIdx.x * 256 + threadIdx.x;  // 0 .. 1048575
  int nrow = idx >> 10, k = idx & 1023;
  out[idx] = f2bf(in[k * 1024 + nrow]);
}

// ---------------- GEMM: C[M,N] = A[M,K] * Bt[N,K]^T + bias ----------------
template <int OUTF32>
__global__ __launch_bounds__(256) void gemm_bf16_kernel(const u16* __restrict__ A,
                                                        const u16* __restrict__ Bt,
                                                        const float* __restrict__ bias,
                                                        void* __restrict__ Cout,
                                                        int M, int N, int K) {
  __shared__ __align__(16) u16 lA[128 * 32];
  __shared__ __align__(16) u16 lB[128 * 32];
  const int tid = threadIdx.x, lane = tid & 31, wid = tid >> 5;
  const int wm = wid >> 1, wn = wid & 1;                 // 4 x 2 wave grid
  const int bM = blockIdx.x * 128, bN = blockIdx.y * 128;
  v8f acc[2][4];
#pragma unroll
  for (int i = 0; i < 2; i++)
#pragma unroll
    for (int j = 0; j < 4; j++) acc[i][j] = (v8f){0, 0, 0, 0, 0, 0, 0, 0};

  for (int k0 = 0; k0 < K; k0 += 32) {
#if defined(USE_TDM)
    // One wave drives the Tensor Data Mover for both tiles, waits TENSORcnt, then the
    // block barrier releases everyone.
    if (wid == 0) {
      u32 ldsA = (u32)(size_t)(void*)&lA[0];
      u32 ldsB = (u32)(size_t)(void*)&lB[0];
      u64 gA = (u64)(size_t)&A[(size_t)bM * K + k0];
      u64 gB = (u64)(size_t)&Bt[(size_t)bN * K + k0];
      tdm_load_2d(ldsA, gA, (u32)K, (u32)M, 32u, 128u, (u64)K);
      tdm_load_2d(ldsB, gB, (u32)K, (u32)N, 32u, 128u, (u64)K);
      __builtin_amdgcn_s_wait_tensorcnt(0);
    }
    __syncthreads();
#else
#pragma unroll
    for (int i = 0; i < 2; i++) {
      int idx = tid + i * 256;                 // 512 16B chunks per tile
      int row = idx >> 2, cc = (idx & 3) * 8;
      *(uint4*)&lA[row * 32 + cc] = *(const uint4*)&A[(size_t)(bM + row) * K + k0 + cc];
      *(uint4*)&lB[row * 32 + cc] = *(const uint4*)&Bt[(size_t)(bN + row) * K + k0 + cc];
    }
    __syncthreads();
#endif
    v16bf af[2], bfr[4];
#pragma unroll
    for (int i = 0; i < 2; i++) af[i] = ld_frag(&lA[(wm * 32 + i * 16) * 32], 32, lane);
#pragma unroll
    for (int j = 0; j < 4; j++) bfr[j] = ld_frag(&lB[(wn * 64 + j * 16) * 32], 32, lane);
#pragma unroll
    for (int i = 0; i < 2; i++)
#pragma unroll
      for (int j = 0; j < 4; j++) acc[i][j] = wmma_bf16(af[i], bfr[j], acc[i][j]);
    __syncthreads();
  }

  const int rowAdd = (lane & 16) ? 8 : 0;
  const int nIdx = lane & 15;
#pragma unroll
  for (int i = 0; i < 2; i++)
#pragma unroll
    for (int j = 0; j < 4; j++)
#pragma unroll
      for (int r = 0; r < 8; r++) {
        int row = bM + wm * 32 + i * 16 + r + rowAdd;
        int col = bN + wn * 64 + j * 16 + nIdx;
        float v = acc[i][j][r] + bias[col];
        if (OUTF32) ((float*)Cout)[(size_t)row * N + col] = v;
        else        ((u16*)Cout)[(size_t)row * N + col] = f2bf(v);
      }
}

// ---------------- sliding-window flash attention ----------------
// B=2,H=16,S=2048,D=1024,hd=64,window=128. One wave per 16-row query tile per (b,h).
__global__ __launch_bounds__(256) void swattn_kernel(const u16* __restrict__ Q,
                                                     const u16* __restrict__ Kb,
                                                     const u16* __restrict__ Vb,
                                                     u16* __restrict__ AB,
                                                     float* __restrict__ AW) {
  __shared__ __align__(16) u16 lP[8][16 * 32];
  const int tid = threadIdx.x, lane = tid & 31, wid = tid >> 5;
  int gw = blockIdx.x * 8 + wid;            // 0..4095
  int qt = gw & 127, h = (gw >> 7) & 15, b = gw >> 11;
  int q0 = qt * 16;
  const size_t rb = (size_t)b * 2048 * 1024;
  const int hc = h * 64;
  const int rowAdd = (lane & 16) ? 8 : 0;
  const int nIdx = lane & 15;

  v16bf aQ0 = ld_frag(Q + rb + (size_t)q0 * 1024 + hc, 1024, lane);       // d 0..31
  v16bf aQ1 = ld_frag(Q + rb + (size_t)q0 * 1024 + hc + 32, 1024, lane);  // d 32..63

  float m_i[8], l_i[8];
  v8f o[4];
  v8f pst[5][2];   // unnormalized P kept for attn-weights output
#pragma unroll
  for (int r = 0; r < 8; r++) { m_i[r] = -1e30f; l_i[r] = 0.f; }
#pragma unroll
  for (int t = 0; t < 4; t++) o[t] = (v8f){0, 0, 0, 0, 0, 0, 0, 0};

#pragma unroll
  for (int pp = 0; pp < 5; pp++) {
    int k0 = q0 - 144 + 32 * pp;   // covers keys [q0-144, q0+15] >= window [q0-127, q0+15]
    v16bf bk00 = ld_frag_clamp(Kb + rb, k0,      1024, hc,      lane, 2047);
    v16bf bk01 = ld_frag_clamp(Kb + rb, k0,      1024, hc + 32, lane, 2047);
    v16bf bk10 = ld_frag_clamp(Kb + rb, k0 + 16, 1024, hc,      lane, 2047);
    v16bf bk11 = ld_frag_clamp(Kb + rb, k0 + 16, 1024, hc + 32, lane, 2047);
    v8f z = (v8f){0, 0, 0, 0, 0, 0, 0, 0};
    v8f s0 = wmma_bf16(aQ1, bk01, wmma_bf16(aQ0, bk00, z));   // keys k0..k0+15
    v8f s1 = wmma_bf16(aQ1, bk11, wmma_bf16(aQ0, bk10, z));   // keys k0+16..k0+31

    float fac[8], pc0[8], pc1[8];
#pragma unroll
    for (int r = 0; r < 8; r++) {
      int i = q0 + r + rowAdd;
      int j0 = k0 + nIdx, j1 = k0 + 16 + nIdx;
      float x0 = (j0 >= 0 && j0 <= i && j0 > i - 128) ? s0[r] * 0.125f : -1e30f;
      float x1 = (j1 >= 0 && j1 <= i && j1 > i - 128) ? s1[r] * 0.125f : -1e30f;
      float mx = redmax16(fmaxf(x0, x1));
      float mn = fmaxf(m_i[r], mx);
      float fr = __expf(m_i[r] - mn);
      float p0 = __expf(x0 - mn), p1 = __expf(x1 - mn);
      float rs = redsum16(p0 + p1);
      l_i[r] = l_i[r] * fr + rs;
      m_i[r] = mn;
      fac[r] = fr; pc0[r] = p0; pc1[r] = p1;
    }
#pragma unroll
    for (int t = 0; t < 4; t++)
#pragma unroll
      for (int r = 0; r < 8; r++) o[t][r] *= fac[r];
#pragma unroll
    for (int prev = 0; prev < pp; prev++)
#pragma unroll
      for (int t = 0; t < 2; t++)
#pragma unroll
        for (int r = 0; r < 8; r++) pst[prev][t][r] *= fac[r];
#pragma unroll
    for (int r = 0; r < 8; r++) { pst[pp][0][r] = pc0[r]; pst[pp][1][r] = pc1[r]; }

    // C-layout -> A-layout via LDS round trip (bf16, 16x32 row-major)
    u16* lp = &lP[wid][0];
#pragma unroll
    for (int r = 0; r < 8; r++) {
      int row = r + rowAdd;
      lp[row * 32 + nIdx]      = f2bf(pc0[r]);
      lp[row * 32 + 16 + nIdx] = f2bf(pc1[r]);
    }
    asm volatile("s_wait_dscnt 0" ::: "memory");
    v16bf aP = ld_frag(lp, 32, lane);
#pragma unroll
    for (int t = 0; t < 4; t++) {
      v16bf bv = ld_vfrag(Vb + rb, k0, hc + 16 * t + nIdx, lane);
      o[t] = wmma_bf16(aP, bv, o[t]);
    }
  }

  float inv[8];
#pragma unroll
  for (int r = 0; r < 8; r++) inv[r] = 1.f / l_i[r];
#pragma unroll
  for (int t = 0; t < 4; t++)
#pragma unroll
    for (int r = 0; r < 8; r++) {
      int row = r + rowAdd;
      AB[rb + (size_t)(q0 + row) * 1024 + hc + 16 * t + nIdx] = f2bf(o[t][r] * inv[r]);
    }

  // dense attention-weights output (window entries; remainder pre-zeroed by memset)
  size_t wbase = ((size_t)(b * 16 + h)) * 2048 * 2048;
#pragma unroll
  for (int pp = 0; pp < 5; pp++) {
    int k0 = q0 - 144 + 32 * pp;
#pragma unroll
    for (int t = 0; t < 2; t++) {
      int j = k0 + 16 * t + nIdx;
      if (j >= 0) {
#pragma unroll
        for (int r = 0; r < 8; r++) {
          int i = q0 + r + rowAdd;
          AW[wbase + (size_t)i * 2048 + j] = pst[pp][t][r] * inv[r];
        }
      }
    }
  }
}

extern "C" void kernel_launch(void* const* d_in, const int* in_sizes, int n_in,
                              void* d_out, int out_size, void* d_ws, size_t ws_size,
                              hipStream_t stream) {
  const float* x  = (const float*)d_in[0];
  const float* Wq = (const float*)d_in[1];
  const float* bq = (const float*)d_in[2];
  const float* Wk = (const float*)d_in[3];
  const float* bk = (const float*)d_in[4];
  const float* Wv = (const float*)d_in[5];
  const float* bv = (const float*)d_in[6];
  const float* Wo = (const float*)d_in[7];
  const float* bo = (const float*)d_in[8];
  // d_in[9]=num_heads(16), d_in[10]=window(128): fixed constants baked in.

  const int M = 4096, N = 1024, K = 1024;       // B*S=4096, D=1024
  u16* ws  = (u16*)d_ws;
  u16* xb  = ws;                  // 4096x1024 bf16
  u16* wqT = xb  + 4194304;       // 1024x1024 bf16 (transposed)
  u16* wkT = wqT + 1048576;
  u16* wvT = wkT + 1048576;
  u16* woT = wvT + 1048576;
  u16* qb  = woT + 1048576;       // 4096x1024 bf16
  u16* kb  = qb  + 4194304;
  u16* vb  = kb  + 4194304;
  u16* ab  = vb  + 4194304;       // attention output bf16
  float* out   = (float*)d_out;               // [2,2048,1024] f32
  float* attnW = out + 4194304;               // [2,16,2048,2048] f32

  f32_to_bf16_kernel<<<16384, 256, 0, stream>>>(x, xb, 4194304);
  transpose_to_bf16_kernel<<<4096, 256, 0, stream>>>(Wq, wqT);
  transpose_to_bf16_kernel<<<4096, 256, 0, stream>>>(Wk, wkT);
  transpose_to_bf16_kernel<<<4096, 256, 0, stream>>>(Wv, wvT);
  transpose_to_bf16_kernel<<<4096, 256, 0, stream>>>(Wo, woT);

  dim3 gg(M / 128, N / 128);
  gemm_bf16_kernel<0><<<gg, 256, 0, stream>>>(xb, wqT, bq, qb, M, N, K);
  gemm_bf16_kernel<0><<<gg, 256, 0, stream>>>(xb, wkT, bk, kb, M, N, K);
  gemm_bf16_kernel<0><<<gg, 256, 0, stream>>>(xb, wvT, bv, vb, M, N, K);

  (void)hipMemsetAsync(attnW, 0, (size_t)134217728 * sizeof(float), stream);
  swattn_kernel<<<512, 256, 0, stream>>>(qb, kb, vb, ab, attnW);

  gemm_bf16_kernel<1><<<gg, 256, 0, stream>>>(ab, woT, bo, out, M, N, K);
}
